// Net_68075231642255
// MI455X (gfx1250) — compile-verified, matrix operand order checked by reading, not run on hardware
//
#include <hip/hip_runtime.h>
#include <cstddef>
#include <cstdint>

typedef __attribute__((ext_vector_type(16))) _Float16 v16h;
typedef __attribute__((ext_vector_type(8)))  _Float16 v8h;
typedef __attribute__((ext_vector_type(8)))  float    v8f;
typedef __attribute__((ext_vector_type(4)))  int      v4i;

#define WGSZ 256

// ---------- monotonic float <-> uint key (for atomic segment-max) ----------
__device__ __forceinline__ unsigned f2key(float f) {
  unsigned b = __float_as_uint(f);
  return (b & 0x80000000u) ? ~b : (b | 0x80000000u);
}
__device__ __forceinline__ float key2f(unsigned k) {
  unsigned b = (k & 0x80000000u) ? (k ^ 0x80000000u) : ~k;
  return __uint_as_float(b);
}

// ---------- generic fill ----------
__global__ void fill_u32(unsigned* __restrict__ p, unsigned v, long long n) {
  long long i = (long long)blockIdx.x * blockDim.x + threadIdx.x;
  if (i < n) p[i] = v;
}

// ---------- operand packing: f32 -> zero-padded f16 ----------
__global__ void pack_a(const float* __restrict__ A, _Float16* __restrict__ Ap,
                       int M, int K, int Mp, int Kp) {
  long long i = (long long)blockIdx.x * blockDim.x + threadIdx.x;
  if (i >= (long long)Mp * Kp) return;
  int r = (int)(i / Kp), k = (int)(i % Kp);
  float v = (r < M && k < K) ? A[(size_t)r * K + k] : 0.0f;
  Ap[i] = (_Float16)v;
}
// B[K,Ncol] -> Bt[Np,Kp]  (transposed: Bt[n*Kp+k] = B[k*Ncol+n])
__global__ void pack_bt(const float* __restrict__ B, _Float16* __restrict__ Bt,
                        int K, int Ncol, int Kp, int Np) {
  long long i = (long long)blockIdx.x * blockDim.x + threadIdx.x;
  if (i >= (long long)Np * Kp) return;
  int n = (int)(i / Kp), k = (int)(i % Kp);
  float v = (n < Ncol && k < K) ? B[(size_t)k * Ncol + n] : 0.0f;
  Bt[i] = (_Float16)v;
}

// ---------- async global -> LDS copy (CDNA5, tracked by ASYNCcnt) ----------
__device__ __forceinline__ void async_cp_b128(const _Float16* gsrc, _Float16* ldst) {
  __builtin_amdgcn_global_load_async_to_lds_b128(
      (__attribute__((address_space(1))) v4i*)gsrc,
      (__attribute__((address_space(3))) v4i*)ldst, 0, 0);
}

// ---------- WMMA GEMM: C[M,Ncols] = A @ B (+bias, ReLU) ----------
// Packed f16 operands; 64x64 block tile; 4 waves; wave computes 16x64.
// Double-buffered LDS staging via GLOBAL_LOAD_ASYNC_TO_LDS_B128 (ASYNCcnt),
// 4x v_wmma_f32_16x16x32_f16 per k-step overlapped with next tile's DMA.
__global__ void __launch_bounds__(128) gemm_wmma(
    const _Float16* __restrict__ Ap, const _Float16* __restrict__ Bt,
    const float* __restrict__ bias, float* __restrict__ C,
    int M, int Kp, int Ncols, int epilogue) {
  __shared__ _Float16 As[2][64 * 32];  // [row][k]  k-contiguous
  __shared__ _Float16 Bs[2][64 * 32];  // [col][k]  k-contiguous (from Bt)
  const int tid  = threadIdx.x;
  const int wave = tid >> 5;
  const int lane = tid & 31;
  const int hg   = lane >> 4;
  const int lr   = lane & 15;
  const int m0   = blockIdx.x * 64;
  const int n0   = blockIdx.y * 64;

  v8f acc[4] = {{}, {}, {}, {}};

  // stage one 64x32 A-tile + 64x32 B-tile into LDS buffer `buf` (4 async ops/thread)
  auto stage = [&](int buf, int k0) {
#pragma unroll
    for (int it = 0; it < 2; ++it) {
      int idx = tid + it * 128;          // v8h index; tile[idx*8] == tile[row*32+ko]
      int row = idx >> 2;
      int ko  = (idx & 3) << 3;
      async_cp_b128(&Ap[(size_t)(m0 + row) * Kp + k0 + ko], &As[buf][idx * 8]);
      async_cp_b128(&Bt[(size_t)(n0 + row) * Kp + k0 + ko], &Bs[buf][idx * 8]);
    }
  };

  const int ktiles = Kp >> 5;
  stage(0, 0);
  for (int kt = 0; kt < ktiles; ++kt) {
    const int buf = kt & 1;
    if (kt + 1 < ktiles) {
      stage(buf ^ 1, (kt + 1) << 5);     // prefetch next tile (stays in flight)
      asm volatile("s_wait_asynccnt 0x4" ::: "memory");  // retire this tile's 4 ops
    } else {
      asm volatile("s_wait_asynccnt 0x0" ::: "memory");
    }
    __syncthreads();

    // A fragment: element i -> K = (i<8 ? i : 16+(i-8)) + 8*hg
    const int arow = wave * 16 + lr;
    v8h al = *(const v8h*)&As[buf][arow * 32 + 8 * hg];
    v8h ah = *(const v8h*)&As[buf][arow * 32 + 16 + 8 * hg];
    v16h af;
#pragma unroll
    for (int i = 0; i < 8; ++i) { af[i] = al[i]; af[i + 8] = ah[i]; }

#pragma unroll
    for (int t = 0; t < 4; ++t) {
      const int ncol = t * 16 + lr;
      v8h bl = *(const v8h*)&Bs[buf][ncol * 32 + 8 * hg];
      v8h bh = *(const v8h*)&Bs[buf][ncol * 32 + 16 + 8 * hg];
      v16h bf;
#pragma unroll
      for (int i = 0; i < 8; ++i) { bf[i] = bl[i]; bf[i + 8] = bh[i]; }
      acc[t] = __builtin_amdgcn_wmma_f32_16x16x32_f16(false, af, false, bf,
                                                      (short)0, acc[t], false, false);
    }
    __syncthreads();
  }

  // C layout: VGPR r -> row (r + 8*hg), col = lr
#pragma unroll
  for (int t = 0; t < 4; ++t) {
#pragma unroll
    for (int r = 0; r < 8; ++r) {
      int gm = m0 + wave * 16 + r + 8 * hg;
      int gn = n0 + t * 16 + lr;
      if (gm < M && gn < Ncols) {
        float v = acc[t][r];
        if (epilogue == 1) { v += bias[gn]; v = v > 0.0f ? v : 0.0f; }
        C[(size_t)gm * Ncols + gn] = v;
      }
    }
  }
}

// ---------- a[n,h] = sum_c xw[n,h,c] * att[h,c] ----------
__global__ void att_proj(const float* __restrict__ xw, const float* __restrict__ att,
                         float* __restrict__ out, int n, int H, int C) {
  long long i = (long long)blockIdx.x * blockDim.x + threadIdx.x;
  if (i >= (long long)n * H) return;
  int node = (int)(i / H), h = (int)(i % H);
  const float* row = xw + (size_t)node * H * C + h * C;
  const float* a   = att + h * C;
  float s = 0.0f;
  for (int c = 0; c < C; ++c) s += row[c] * a[c];
  out[i] = s;
}

__device__ __forceinline__ void edge_decode(const int* ei, int E, long long e,
                                            int& s, int& d) {
  if (e < E) { s = ei[e]; d = ei[(size_t)E + e]; }
  else       { s = (int)(e - E); d = s; }           // self-loops
}

__global__ void edge_max(const int* __restrict__ ei, int E, int Etot,
                         const float* __restrict__ asrc, const float* __restrict__ adst,
                         unsigned* __restrict__ amaxk, int H) {
  long long i = (long long)blockIdx.x * blockDim.x + threadIdx.x;
  if (i >= (long long)Etot * H) return;
  long long e = i / H; int h = (int)(i % H);
  int s, d; edge_decode(ei, E, e, s, d);
  float al = asrc[(size_t)s * H + h] + adst[(size_t)d * H + h];
  al = al > 0.0f ? al : 0.2f * al;                  // leaky_relu(0.2)
  atomicMax(&amaxk[(size_t)d * H + h], f2key(al));
}

__global__ void edge_sum(const int* __restrict__ ei, int E, int Etot,
                         const float* __restrict__ asrc, const float* __restrict__ adst,
                         const unsigned* __restrict__ amaxk, float* __restrict__ denom, int H) {
  long long i = (long long)blockIdx.x * blockDim.x + threadIdx.x;
  if (i >= (long long)Etot * H) return;
  long long e = i / H; int h = (int)(i % H);
  int s, d; edge_decode(ei, E, e, s, d);
  float al = asrc[(size_t)s * H + h] + adst[(size_t)d * H + h];
  al = al > 0.0f ? al : 0.2f * al;
  float ex = __expf(al - key2f(amaxk[(size_t)d * H + h]));
  atomicAdd(&denom[(size_t)d * H + h], ex);
}

__global__ void edge_aggr(const int* __restrict__ ei, int E, int Etot,
                          const float* __restrict__ asrc, const float* __restrict__ adst,
                          const unsigned* __restrict__ amaxk, const float* __restrict__ denom,
                          const float* __restrict__ xw, float* __restrict__ out, int H, int C) {
  long long i = (long long)blockIdx.x * blockDim.x + threadIdx.x;
  if (i >= (long long)Etot * H) return;
  long long e = i / H; int h = (int)(i % H);
  int s, d; edge_decode(ei, E, e, s, d);
  float al = asrc[(size_t)s * H + h] + adst[(size_t)d * H + h];
  al = al > 0.0f ? al : 0.2f * al;
  float attn = __expf(al - key2f(amaxk[(size_t)d * H + h])) /
               (denom[(size_t)d * H + h] + 1e-16f);
  const int F = H * C;
  const float* src = xw + (size_t)s * F + h * C;
  float* dst = out + (size_t)d * F + h * C;
  for (int c = 0; c < C; ++c) atomicAdd(&dst[c], src[c] * attn);
}

__global__ void bias_elu(float* __restrict__ h, const float* __restrict__ bias,
                         int n, int F) {
  long long i = (long long)blockIdx.x * blockDim.x + threadIdx.x;
  if (i >= (long long)n * F) return;
  float v = h[i] + bias[i % F];
  h[i] = v > 0.0f ? v : (__expf(v) - 1.0f);
}

// ---------- BatchNorm (train-mode batch stats): one block per channel ----------
__global__ void bn_stats(const float* __restrict__ h, int n, int C,
                         float* __restrict__ mu, float* __restrict__ rsig) {
  const int c = blockIdx.x;
  float s = 0.0f, s2 = 0.0f;
  for (int r = threadIdx.x; r < n; r += blockDim.x) {
    float v = h[(size_t)r * C + c]; s += v; s2 += v * v;
  }
  __shared__ float ss[256], ss2[256];
  ss[threadIdx.x] = s; ss2[threadIdx.x] = s2;
  __syncthreads();
  for (int off = 128; off > 0; off >>= 1) {
    if (threadIdx.x < off) { ss[threadIdx.x] += ss[threadIdx.x + off];
                             ss2[threadIdx.x] += ss2[threadIdx.x + off]; }
    __syncthreads();
  }
  if (threadIdx.x == 0) {
    float m = ss[0] / n;
    float var = ss2[0] / n - m * m;
    mu[c] = m; rsig[c] = rsqrtf(var + 1e-5f);
  }
}

__global__ void bn_apply(const float* __restrict__ in, float* __restrict__ out,
                         int n, int C, int ostride,
                         const float* __restrict__ mu, const float* __restrict__ rsig,
                         const float* __restrict__ gamma, const float* __restrict__ beta) {
  long long i = (long long)blockIdx.x * blockDim.x + threadIdx.x;
  if (i >= (long long)n * C) return;
  int r = (int)(i / C), c = (int)(i % C);
  float v = in[(size_t)r * C + c];
  out[(size_t)r * ostride + c] = gamma[c] * (v - mu[c]) * rsig[c] + beta[c];
}

// ---------- final head layer: N x 32 @ 32 x 1 + bias + ReLU ----------
__global__ void head_final(const float* __restrict__ h, const float* __restrict__ W,
                           const float* __restrict__ b, float* __restrict__ col,
                           int n, int K) {
  long long i = (long long)blockIdx.x * blockDim.x + threadIdx.x;
  if (i >= n) return;
  float s = b[0];
  const float* row = h + (size_t)i * K;
  for (int k = 0; k < K; ++k) s += row[k] * W[k];
  col[i] = s > 0.0f ? s : 0.0f;
}

static inline unsigned cdiv(long long a, long long b) { return (unsigned)((a + b - 1) / b); }
static inline int rup(int a, int b) { return ((a + b - 1) / b) * b; }

extern "C" void kernel_launch(void* const* d_in, const int* in_sizes, int n_in,
                              void* d_out, int out_size, void* d_ws, size_t ws_size,
                              hipStream_t stream) {
  const float* x  = (const float*)d_in[0];
  const int*   ei = (const int*)d_in[1];
  const int N    = in_sizes[0] / 32;
  const int E    = in_sizes[1] / 2;
  const int Etot = E + N;
  const int Mp   = rup(N, 64);

  // workspace layout (4-byte units unless noted)
  float*    P0    = (float*)d_ws;                       // N*256 ping
  float*    P1    = P0 + (size_t)N * 256;               // N*256 pong
  float*    P2    = P1 + (size_t)N * 256;               // N*144 xw / head scratch
  unsigned* amaxk = (unsigned*)(P2 + (size_t)N * 144);  // N*12
  float*    denom = (float*)(amaxk + (size_t)N * 12);   // N*12
  float*    asrc  = denom + (size_t)N * 12;             // N*12
  float*    adst  = asrc + (size_t)N * 12;              // N*12
  float*    mu    = adst + (size_t)N * 12;              // 256
  float*    rsig  = mu + 256;                           // 256
  _Float16* Ap    = (_Float16*)(rsig + 256);            // Mp*256 halfs (packed A)
  _Float16* Bt    = Ap + (size_t)Mp * 256;              // 256*256 halfs (packed B^T)

  const unsigned NEGINF_KEY = 0x007FFFFFu;              // f2key(-inf)

  auto gemm = [&](const float* A, const float* B, const float* bias, float* Cc,
                  int M, int K, int Ncols, int epi) {
    const int Kp = rup(K, 32), Np = rup(Ncols, 64), Mpp = rup(M, 64);
    pack_a <<<cdiv((long long)Mpp * Kp, WGSZ), WGSZ, 0, stream>>>(A, Ap, M, K, Mpp, Kp);
    pack_bt<<<cdiv((long long)Np * Kp, WGSZ), WGSZ, 0, stream>>>(B, Bt, K, Ncols, Kp, Np);
    dim3 g(Mpp / 64, Np / 64);
    gemm_wmma<<<g, dim3(128), 0, stream>>>(Ap, Bt, bias, Cc, M, Kp, Ncols, epi);
  };

  // ----- GAT layer: out = elu(segment_softmax_aggregate(hin@W) + bias) -----
  auto gat = [&](const float* hin, int K, int H, int C,
                 const float* W, const float* attS, const float* attD,
                 const float* bias, float* out) {
    const int F = H * C;
    gemm(hin, W, nullptr, P2, N, K, F, 0);                                   // xw
    att_proj<<<cdiv((long long)N * H, WGSZ), WGSZ, 0, stream>>>(P2, attS, asrc, N, H, C);
    att_proj<<<cdiv((long long)N * H, WGSZ), WGSZ, 0, stream>>>(P2, attD, adst, N, H, C);
    fill_u32<<<cdiv((long long)N * H, WGSZ), WGSZ, 0, stream>>>(amaxk, NEGINF_KEY, (long long)N * H);
    fill_u32<<<cdiv((long long)N * H, WGSZ), WGSZ, 0, stream>>>((unsigned*)denom, 0u, (long long)N * H);
    fill_u32<<<cdiv((long long)N * F, WGSZ), WGSZ, 0, stream>>>((unsigned*)out, 0u, (long long)N * F);
    const long long eh = (long long)Etot * H;
    edge_max <<<cdiv(eh, WGSZ), WGSZ, 0, stream>>>(ei, E, Etot, asrc, adst, amaxk, H);
    edge_sum <<<cdiv(eh, WGSZ), WGSZ, 0, stream>>>(ei, E, Etot, asrc, adst, amaxk, denom, H);
    edge_aggr<<<cdiv(eh, WGSZ), WGSZ, 0, stream>>>(ei, E, Etot, asrc, adst, amaxk, denom, P2, out, H, C);
    bias_elu <<<cdiv((long long)N * F, WGSZ), WGSZ, 0, stream>>>(out, bias, N, F);
  };

  // ----- Linear + ReLU + BatchNorm (in-place BN) -----
  auto lin_bn = [&](const float* in, float* out, int K, int C, int widx) {
    gemm(in, (const float*)d_in[widx], (const float*)d_in[widx + 1], out, N, K, C, 1);
    bn_stats<<<C, 256, 0, stream>>>(out, N, C, mu, rsig);
    bn_apply<<<cdiv((long long)N * C, WGSZ), WGSZ, 0, stream>>>(
        out, out, N, C, C, mu, rsig,
        (const float*)d_in[widx + 2], (const float*)d_in[widx + 3]);
  };

  // GAT1: x[N,32] -> P0[N,144]
  gat(x, 32, 12, 12, (const float*)d_in[2], (const float*)d_in[3],
      (const float*)d_in[4], (const float*)d_in[5], P0);
  // GAT2: P0[N,144] -> P1[N,64]
  gat(P0, 144, 8, 8, (const float*)d_in[6], (const float*)d_in[7],
      (const float*)d_in[8], (const float*)d_in[9], P1);

  // MLP 64->128->256->128->64 (ping-pong P1 <-> P0)
  lin_bn(P1, P0, 64, 128, 10);
  lin_bn(P0, P1, 128, 256, 14);
  lin_bn(P1, P0, 256, 128, 18);
  lin_bn(P0, P1, 128, 64, 22);

  // 3 heads: 64->32->32->1, each layer Linear+ReLU+BN; concat to d_out[N,3]
  float* outp = (float*)d_out;
  for (int j = 0; j < 3; ++j) {
    const int b = 26 + j * 12;
    lin_bn(P1, P2, 64, 32, b);        // head L1 -> P2
    lin_bn(P2, P0, 32, 32, b + 4);    // head L2 -> P0
    head_final<<<cdiv(N, WGSZ), WGSZ, 0, stream>>>(
        P0, (const float*)d_in[b + 8], (const float*)d_in[b + 9], P2, N, 32);
    bn_stats<<<1, 256, 0, stream>>>(P2, N, 1, mu, rsig);
    bn_apply<<<cdiv(N, WGSZ), WGSZ, 0, stream>>>(
        P2, outp + j, N, 1, 3, mu, rsig,
        (const float*)d_in[b + 10], (const float*)d_in[b + 11]);
  }
}
